// KnowledgeFeatureFusion_89687507075178
// MI455X (gfx1250) — compile-verified, matrix operand order checked by reading, not run on hardware
//
#include <hip/hip_runtime.h>
#include <math.h>

// ---------------------------------------------------------------------------
// Problem constants (from reference)
// ---------------------------------------------------------------------------
#define NN      512          // NUM_NODES
#define GH      32           // GNN_HIDDEN
#define NHEAD   4
#define HD      8            // head dim
#define EDGES   4096
#define BPT     512          // B*P*T = 8*4*16
#define NQ      (BPT * 64)   // 32768 query rows
#define CHI     128
#define CLO     64
#define KQ      (CHI + CLO)        // 192
#define KF      (CHI + CLO + GH)   // 224
#define COUT    128

typedef __attribute__((ext_vector_type(16))) __bf16 v16bf;
typedef __attribute__((ext_vector_type(8)))  __bf16 v8bf;
typedef __attribute__((ext_vector_type(4)))  __bf16 v4bf;
typedef __attribute__((ext_vector_type(8)))  float  v8f;

static __device__ inline __bf16 f2bf(float f) { return (__bf16)f; }

static __device__ inline v16bf cat8(v8bf a, v8bf b) {
  return __builtin_shufflevector(a, b, 0, 1, 2, 3, 4, 5, 6, 7,
                                       8, 9, 10, 11, 12, 13, 14, 15);
}

static __device__ inline v8f wmma_bf16(v16bf a, v16bf b, v8f c) {
  return __builtin_amdgcn_wmma_f32_16x16x32_bf16(
      false, a, false, b, (short)0, c, false, false);
}

// Element-wise fragment builder (only used by the tiny GCN GEMMs).
// CDNA5 ISA 7.12.2 striping: lanes 0-15 K={0..7,16..23}; lanes 16-31 K={8..15,24..31}
template <typename F>
static __device__ inline v16bf make_frag(F f, int lane) {
  const int rc = lane & 15;
  const int kb = (lane >> 4) << 3;
  v16bf a;
#pragma unroll
  for (int e = 0; e < 16; ++e) {
    int k = kb + (e < 8 ? e : e + 8);
    a[e] = f2bf(f(rc, k));
  }
  return a;
}

// ---------------------------------------------------------------------------
// Prep: fold Wq_proj*Wq (+bias) into Wqc/bqc; build Wcomb = [Wf[0:192]; Wo*Wf[192:224]]
// and bias bfc = bo*Wf[192:224] + bf
// ---------------------------------------------------------------------------
__global__ void prep_weights_kernel(const float* __restrict__ Wq_proj,
                                    const float* __restrict__ bq_proj,
                                    const float* __restrict__ Wq,
                                    const float* __restrict__ bq,
                                    const float* __restrict__ Wo,
                                    const float* __restrict__ bo,
                                    const float* __restrict__ Wf,
                                    const float* __restrict__ bf,
                                    float* __restrict__ Wqc,    // [192,32]
                                    float* __restrict__ bqc,    // [32]
                                    float* __restrict__ Wcomb,  // [224,128]
                                    float* __restrict__ bfc) {  // [128]
  const int tid = threadIdx.x;
  for (int i = tid; i < KQ * GH; i += blockDim.x) {
    int r = i >> 5, c = i & 31;
    float s = 0.f;
    for (int l = 0; l < GH; ++l) s += Wq_proj[r * GH + l] * Wq[l * GH + c];
    Wqc[i] = s;
  }
  if (tid < GH) {
    float s = bq[tid];
    for (int l = 0; l < GH; ++l) s += bq_proj[l] * Wq[l * GH + tid];
    bqc[tid] = s;
  }
  for (int i = tid; i < KQ * COUT; i += blockDim.x)   // rows 0..191: copy Wf
    Wcomb[i] = Wf[i];
  for (int i = tid; i < GH * COUT; i += blockDim.x) { // rows 192..223: Wo @ Wf_o
    int r = i >> 7, c = i & 127;
    float s = 0.f;
    for (int l = 0; l < GH; ++l) s += Wo[r * GH + l] * Wf[(KQ + l) * COUT + c];
    Wcomb[(KQ + r) * COUT + c] = s;
  }
  if (tid < COUT) {
    float s = bf[tid];
    for (int l = 0; l < GH; ++l) s += bo[l] * Wf[(KQ + l) * COUT + tid];
    bfc[tid] = s;
  }
}

// ---------------------------------------------------------------------------
// Pack a row-major f32 weight W[K x n] into per-wave bf16 B-fragments:
// dst[((kc*ntiles + nt)*32 + lane)*16 + e]  (lane loads 32B contiguously)
// ---------------------------------------------------------------------------
__global__ void pack_b_kernel(const float* __restrict__ W, int n, int nchunks,
                              __bf16* __restrict__ dst) {
  int t = blockIdx.x * blockDim.x + threadIdx.x;
  int ntiles = n >> 4;
  if (t >= nchunks * ntiles * 32) return;
  int lane = t & 31;
  int frag = t >> 5;
  int nt = frag % ntiles;
  int kc = frag / ntiles;
  int col = nt * 16 + (lane & 15);
  int kb = kc * 32 + ((lane >> 4) << 3);
  __bf16* o = dst + (size_t)t * 16;
#pragma unroll
  for (int e = 0; e < 16; ++e) {
    int k = kb + (e < 8 ? e : e + 8);
    o[e] = f2bf(W[k * n + col]);
  }
}

// K fragments for score WMMA: dst[((h*32 + c)*32 + lane)*16 + e], zero-padded K>=8
__global__ void pack_k_kernel(const float* __restrict__ kbuf,
                              __bf16* __restrict__ dst) {
  int t = blockIdx.x * blockDim.x + threadIdx.x;
  if (t >= NHEAD * (NN / 16) * 32) return;
  int lane = t & 31;
  int frag = t >> 5;
  int c = frag & 31;
  int h = frag >> 5;
  int col = lane & 15;
  int kb = (lane >> 4) << 3;
  __bf16* o = dst + (size_t)t * 16;
#pragma unroll
  for (int e = 0; e < 16; ++e) {
    int k = kb + (e < 8 ? e : e + 8);
    o[e] = (k < HD) ? f2bf(kbuf[(c * 16 + col) * GH + h * HD + k]) : (__bf16)0.0f;
  }
}

// V fragments for attn*V WMMA: dst[((h*16 + c)*32 + lane)*16 + e], cols>=8 zero
__global__ void pack_v_kernel(const float* __restrict__ vbuf,
                              __bf16* __restrict__ dst) {
  int t = blockIdx.x * blockDim.x + threadIdx.x;
  if (t >= NHEAD * (NN / 32) * 32) return;
  int lane = t & 31;
  int frag = t >> 5;
  int c = frag & 15;
  int h = frag >> 4;
  int d = lane & 15;
  int kb = (lane >> 4) << 3;
  __bf16* o = dst + (size_t)t * 16;
#pragma unroll
  for (int e = 0; e < 16; ++e) {
    int key = c * 32 + kb + (e < 8 ? e : e + 8);
    o[e] = (d < HD) ? f2bf(vbuf[key * GH + h * HD + d]) : (__bf16)0.0f;
  }
}

// f32 -> bf16 row-major convert, 4 elems/thread (vector load + 8B store)
__global__ void cvt_bf4_kernel(const float* __restrict__ s,
                               __bf16* __restrict__ d, int n4) {
  int i = blockIdx.x * blockDim.x + threadIdx.x;
  if (i >= n4) return;
  float4 v = ((const float4*)s)[i];
  v4bf o;
  o[0] = f2bf(v.x); o[1] = f2bf(v.y); o[2] = f2bf(v.z); o[3] = f2bf(v.w);
  ((v4bf*)d)[i] = o;
}

// ---------------------------------------------------------------------------
// GCN normalization helpers
// ---------------------------------------------------------------------------
__global__ void deg_init_kernel(float* __restrict__ deg) {
  int i = blockIdx.x * blockDim.x + threadIdx.x;
  if (i < NN) deg[i] = 1.0f;                       // self loop
}
__global__ void deg_accum_kernel(const int* __restrict__ ei,
                                 float* __restrict__ deg) {
  int e = blockIdx.x * blockDim.x + threadIdx.x;
  if (e < EDGES) atomicAdd(&deg[ei[EDGES + e]], 1.0f);   // dst row
}
__global__ void deg_rsqrt_kernel(float* __restrict__ deg) {
  int i = blockIdx.x * blockDim.x + threadIdx.x;
  if (i < NN) deg[i] = rsqrtf(deg[i]);
}
__global__ void zero_kernel(float* __restrict__ p, int n) {
  int i = blockIdx.x * blockDim.x + threadIdx.x;
  if (i < n) p[i] = 0.0f;
}

// agg[dst] += dinv[src]*dinv[dst] * xw[src]   (edges + self loops)
__global__ void scatter_kernel(const float* __restrict__ xw,
                               const int* __restrict__ ei,
                               const float* __restrict__ dinv,
                               float* __restrict__ agg) {
  int t = blockIdx.x * blockDim.x + threadIdx.x;
  if (t >= (EDGES + NN) * GH) return;
  int e = t >> 5, c = t & 31;
  int s, d;
  if (e < EDGES) { s = ei[e]; d = ei[EDGES + e]; }
  else           { s = d = e - EDGES; }
  float w = dinv[s] * dinv[d];
  atomicAdd(&agg[d * GH + c], w * xw[s * GH + c]);
}

__global__ void finalize_kernel(const float* __restrict__ agg,
                                const float* __restrict__ bias,
                                float* __restrict__ h) {
  int t = blockIdx.x * blockDim.x + threadIdx.x;
  if (t < NN * GH) h[t] = fmaxf(agg[t] + bias[t & 31], 0.0f);
}

// ---------------------------------------------------------------------------
// Tiny generic WMMA GEMM (GCN transforms, K/V projections)
// ---------------------------------------------------------------------------
__global__ void gemm_plain_kernel(const float* __restrict__ A, int lda,
                                  const float* __restrict__ W, int n,
                                  const float* __restrict__ bias,
                                  float* __restrict__ C, int ldc, int K) {
  const int lane = threadIdx.x;
  const int m0 = blockIdx.x * 16;
  const int n0 = blockIdx.y * 16;
  v8f acc = {};
  for (int k0 = 0; k0 < K; k0 += 32) {
    v16bf a = make_frag([&](int r, int k) {
        return A[(m0 + r) * lda + (k0 + k)]; }, lane);
    v16bf b = make_frag([&](int c, int k) {
        return W[(k0 + k) * n + (n0 + c)]; }, lane);
    acc = wmma_bf16(a, b, acc);
  }
  const int col = n0 + (lane & 15);
  const int rb  = m0 + ((lane >> 4) << 3);
  const float bv = bias ? bias[col] : 0.0f;
#pragma unroll
  for (int r = 0; r < 8; ++r) C[(rb + r) * ldc + col] = acc[r] + bv;
}

// ---------------------------------------------------------------------------
// q_bf = bf16( [hi|lo] @ Wqc + bqc )   K=192, N=32, packed operands
// ---------------------------------------------------------------------------
__global__ void gemm_q_kernel(const __bf16* __restrict__ hi_bf,
                              const __bf16* __restrict__ lo_bf,
                              const __bf16* __restrict__ WqcP,
                              const float* __restrict__ bqc,
                              __bf16* __restrict__ q_bf) {
  const int lane = threadIdx.x;
  const int m0 = blockIdx.x * 16;
  const int nt = blockIdx.y;                 // 0..1
  const int row = m0 + (lane & 15);
  const int kb = (lane >> 4) << 3;
  v8f acc = {};
#pragma unroll
  for (int c = 0; c < KQ / 32; ++c) {
    int k0 = c * 32;
    const __bf16* ap = (k0 < CHI)
        ? hi_bf + (size_t)(row >> 6) * CHI + (k0 + kb)
        : lo_bf + (size_t)row * CLO + (k0 - CHI + kb);
    v16bf a = cat8(*(const v8bf*)ap, *(const v8bf*)(ap + 16));
    v16bf b = *(const v16bf*)(WqcP + ((size_t)(c * 2 + nt) * 32 + lane) * 16);
    acc = wmma_bf16(a, b, acc);
  }
  const int col = nt * 16 + (lane & 15);
  const int rb  = m0 + ((lane >> 4) << 3);
  const float bv = bqc[col];
#pragma unroll
  for (int r = 0; r < 8; ++r)
    q_bf[(size_t)(rb + r) * GH + col] = f2bf(acc[r] + bv);
}

// ---------------------------------------------------------------------------
// Attention: one wave per (16-query tile, head); packed Q/K/V fragments.
// Scores (HD=8 zero-padded to K=32) -> LDS softmax -> attn*V, output bf16.
// ---------------------------------------------------------------------------
__global__ void attn_kernel(const __bf16* __restrict__ q_bf,
                            const __bf16* __restrict__ kpack,
                            const __bf16* __restrict__ vpack,
                            __bf16* __restrict__ o_bf) {
  __shared__ float sc[16][NN + 8];
  const int lane = threadIdx.x;
  const int q0 = blockIdx.x * 16;
  const int h  = blockIdx.y;
  const float scale = 0.35355339059327373f;   // 1/sqrt(8)

  // Q fragment: lanes 0-15 hold the head's 8 dims (one 16B load), rest zero
  v8bf z8 = {};
  v8bf qv = z8;
  if (lane < 16)
    qv = *(const v8bf*)(q_bf + (size_t)(q0 + lane) * GH + h * HD);
  v16bf aq = cat8(qv, z8);

  const __bf16* kp = kpack + (size_t)h * (NN / 16) * 32 * 16;
  for (int c = 0; c < NN / 16; ++c) {
    v16bf bk = *(const v16bf*)(kp + ((size_t)c * 32 + lane) * 16);
    v8f s = {};
    s = wmma_bf16(aq, bk, s);
    const int key = c * 16 + (lane & 15);
    const int rb  = (lane >> 4) << 3;
#pragma unroll
    for (int r = 0; r < 8; ++r) sc[rb + r][key] = s[r] * scale;
  }
  __syncthreads();

  if (lane < 16) {
    float m = -1e30f;
    for (int j = 0; j < NN; ++j) m = fmaxf(m, sc[lane][j]);
    float sum = 0.0f;
    for (int j = 0; j < NN; ++j) {
      float e = __expf(sc[lane][j] - m);
      sc[lane][j] = e;
      sum += e;
    }
    float inv = 1.0f / sum;
    for (int j = 0; j < NN; ++j) sc[lane][j] *= inv;
  }
  __syncthreads();

  const int arow = lane & 15;
  const int kb = (lane >> 4) << 3;
  const __bf16* vp = vpack + (size_t)h * (NN / 32) * 32 * 16;
  v8f o = {};
  for (int c = 0; c < NN / 32; ++c) {
    const float* sp = &sc[arow][c * 32 + kb];
    float4 t0 = *(const float4*)sp;
    float4 t1 = *(const float4*)(sp + 4);
    float4 t2 = *(const float4*)(sp + 16);
    float4 t3 = *(const float4*)(sp + 20);
    v16bf aa;
    aa[0]  = f2bf(t0.x); aa[1]  = f2bf(t0.y); aa[2]  = f2bf(t0.z); aa[3]  = f2bf(t0.w);
    aa[4]  = f2bf(t1.x); aa[5]  = f2bf(t1.y); aa[6]  = f2bf(t1.z); aa[7]  = f2bf(t1.w);
    aa[8]  = f2bf(t2.x); aa[9]  = f2bf(t2.y); aa[10] = f2bf(t2.z); aa[11] = f2bf(t2.w);
    aa[12] = f2bf(t3.x); aa[13] = f2bf(t3.y); aa[14] = f2bf(t3.z); aa[15] = f2bf(t3.w);
    v16bf bv = *(const v16bf*)(vp + ((size_t)c * 32 + lane) * 16);
    o = wmma_bf16(aa, bv, o);
  }
  const int d = lane & 15;
  if (d < HD) {
    const int rb = (lane >> 4) << 3;
#pragma unroll
    for (int r = 0; r < 8; ++r)
      o_bf[(size_t)(q0 + rb + r) * GH + h * HD + d] = f2bf(o[r]);
  }
}

// ---------------------------------------------------------------------------
// out = prelu( [hi|lo|o] @ Wcomb + bfc )   K=224, N=128, packed operands
// ---------------------------------------------------------------------------
__global__ void gemm_fused_kernel(const __bf16* __restrict__ hi_bf,
                                  const __bf16* __restrict__ lo_bf,
                                  const __bf16* __restrict__ o_bf,
                                  const __bf16* __restrict__ WfP,
                                  const float* __restrict__ bfc,
                                  const float* __restrict__ prelu_a,
                                  float* __restrict__ out) {
  const int lane = threadIdx.x;
  const int m0 = blockIdx.x * 16;
  const int nt = blockIdx.y;                 // 0..7
  const int row = m0 + (lane & 15);
  const int kb = (lane >> 4) << 3;
  v8f acc = {};
#pragma unroll
  for (int c = 0; c < KF / 32; ++c) {
    int k0 = c * 32;
    const __bf16* ap;
    if (k0 < CHI)      ap = hi_bf + (size_t)(row >> 6) * CHI + (k0 + kb);
    else if (k0 < KQ)  ap = lo_bf + (size_t)row * CLO + (k0 - CHI + kb);
    else               ap = o_bf  + (size_t)row * GH  + (k0 - KQ + kb);
    v16bf a = cat8(*(const v8bf*)ap, *(const v8bf*)(ap + 16));
    v16bf b = *(const v16bf*)(WfP + ((size_t)(c * 8 + nt) * 32 + lane) * 16);
    acc = wmma_bf16(a, b, acc);
  }
  const int col = nt * 16 + (lane & 15);
  const int rb  = m0 + ((lane >> 4) << 3);
  const float alpha = prelu_a[0];
  const float bv = bfc[col];
#pragma unroll
  for (int r = 0; r < 8; ++r) {
    float zv = acc[r] + bv;
    out[(size_t)(rb + r) * COUT + col] = (zv >= 0.0f) ? zv : alpha * zv;
  }
}

// ---------------------------------------------------------------------------
// Launcher
// ---------------------------------------------------------------------------
extern "C" void kernel_launch(void* const* d_in, const int* in_sizes, int n_in,
                              void* d_out, int out_size, void* d_ws, size_t ws_size,
                              hipStream_t stream) {
  const float* hi      = (const float*)d_in[0];
  const float* lo      = (const float*)d_in[1];
  const float* node_x  = (const float*)d_in[2];
  const int*   ei      = (const int*)  d_in[3];
  const float* W1      = (const float*)d_in[4];
  const float* b1      = (const float*)d_in[5];
  const float* W2      = (const float*)d_in[6];
  const float* b2      = (const float*)d_in[7];
  const float* Wq_proj = (const float*)d_in[8];
  const float* bq_proj = (const float*)d_in[9];
  const float* Wq      = (const float*)d_in[10];
  const float* bq      = (const float*)d_in[11];
  const float* Wk      = (const float*)d_in[12];
  const float* bk      = (const float*)d_in[13];
  const float* Wv      = (const float*)d_in[14];
  const float* bv      = (const float*)d_in[15];
  const float* Wo      = (const float*)d_in[16];
  const float* bo      = (const float*)d_in[17];
  const float* Wf      = (const float*)d_in[18];
  const float* bf      = (const float*)d_in[19];
  const float* prelu_a = (const float*)d_in[20];
  float* out = (float*)d_out;

  // ---- workspace carve-up (256B aligned chunks) ----
  char* p = (char*)d_ws;
  auto take = [&](size_t bytes) {
    char* r = p;
    p += (bytes + 255) & ~(size_t)255;
    return r;
  };
  float* deg   = (float*)take(NN * 4);
  float* xw    = (float*)take(NN * GH * 4);
  float* agg   = (float*)take(NN * GH * 4);
  float* hbuf  = (float*)take(NN * GH * 4);
  float* kbuf  = (float*)take(NN * GH * 4);
  float* vbuf  = (float*)take(NN * GH * 4);
  float* Wqc   = (float*)take(KQ * GH * 4);
  float* bqc   = (float*)take(GH * 4);
  float* Wcomb = (float*)take(KF * COUT * 4);
  float* bfc   = (float*)take(COUT * 4);
  __bf16* WqcP  = (__bf16*)take((KQ / 32) * (GH / 16) * 32 * 16 * 2);
  __bf16* WfP   = (__bf16*)take((KF / 32) * (COUT / 16) * 32 * 16 * 2);
  __bf16* kpack = (__bf16*)take((size_t)NHEAD * (NN / 16) * 32 * 16 * 2);
  __bf16* vpack = (__bf16*)take((size_t)NHEAD * (NN / 32) * 32 * 16 * 2);
  __bf16* hi_bf = (__bf16*)take((size_t)BPT * CHI * 2);
  __bf16* lo_bf = (__bf16*)take((size_t)NQ * CLO * 2);
  __bf16* q_bf  = (__bf16*)take((size_t)NQ * GH * 2);
  __bf16* o_bf  = (__bf16*)take((size_t)NQ * GH * 2);

  // ---- fold weights + pack B fragments ----
  prep_weights_kernel<<<1, 256, 0, stream>>>(Wq_proj, bq_proj, Wq, bq,
                                             Wo, bo, Wf, bf,
                                             Wqc, bqc, Wcomb, bfc);
  pack_b_kernel<<<2, 256, 0, stream>>>(Wqc, GH, KQ / 32, WqcP);
  pack_b_kernel<<<7, 256, 0, stream>>>(Wcomb, COUT, KF / 32, WfP);

  // ---- convert A operands to bf16 ----
  cvt_bf4_kernel<<<(BPT * CHI / 4 + 255) / 256, 256, 0, stream>>>(hi, hi_bf, BPT * CHI / 4);
  cvt_bf4_kernel<<<((size_t)NQ * CLO / 4 + 255) / 256, 256, 0, stream>>>(lo, lo_bf, NQ * CLO / 4);

  // ---- gcn_norm: deg -> dinv ----
  deg_init_kernel <<<(NN + 255) / 256, 256, 0, stream>>>(deg);
  deg_accum_kernel<<<(EDGES + 255) / 256, 256, 0, stream>>>(ei, deg);
  deg_rsqrt_kernel<<<(NN + 255) / 256, 256, 0, stream>>>(deg);

  const int nAgg = NN * GH;
  const int nScatter = (EDGES + NN) * GH;

  // ---- GCN layer 1 ----
  gemm_plain_kernel<<<dim3(NN / 16, GH / 16), 32, 0, stream>>>(
      node_x, CLO, W1, GH, nullptr, xw, GH, CLO);
  zero_kernel    <<<(nAgg + 255) / 256, 256, 0, stream>>>(agg, nAgg);
  scatter_kernel <<<(nScatter + 255) / 256, 256, 0, stream>>>(xw, ei, deg, agg);
  finalize_kernel<<<(nAgg + 255) / 256, 256, 0, stream>>>(agg, b1, hbuf);

  // ---- GCN layer 2 ----
  gemm_plain_kernel<<<dim3(NN / 16, GH / 16), 32, 0, stream>>>(
      hbuf, GH, W2, GH, nullptr, xw, GH, GH);
  zero_kernel    <<<(nAgg + 255) / 256, 256, 0, stream>>>(agg, nAgg);
  scatter_kernel <<<(nScatter + 255) / 256, 256, 0, stream>>>(xw, ei, deg, agg);
  finalize_kernel<<<(nAgg + 255) / 256, 256, 0, stream>>>(agg, b2, hbuf);

  // ---- K / V projections + attention fragment packing ----
  gemm_plain_kernel<<<dim3(NN / 16, GH / 16), 32, 0, stream>>>(
      hbuf, GH, Wk, GH, bk, kbuf, GH, GH);
  gemm_plain_kernel<<<dim3(NN / 16, GH / 16), 32, 0, stream>>>(
      hbuf, GH, Wv, GH, bv, vbuf, GH, GH);
  pack_k_kernel<<<(NHEAD * (NN / 16) * 32 + 255) / 256, 256, 0, stream>>>(kbuf, kpack);
  pack_v_kernel<<<(NHEAD * (NN / 32) * 32 + 255) / 256, 256, 0, stream>>>(vbuf, vpack);

  // ---- fused query projection ----
  gemm_q_kernel<<<dim3(NQ / 16, GH / 16), 32, 0, stream>>>(
      hi_bf, lo_bf, WqcP, bqc, q_bf);

  // ---- attention ----
  attn_kernel<<<dim3(NQ / 16, NHEAD), 32, 0, stream>>>(q_bf, kpack, vpack, o_bf);

  // ---- fusion GEMM (+Wo folded) + PReLU ----
  gemm_fused_kernel<<<dim3(NQ / 16, COUT / 16), 32, 0, stream>>>(
      hi_bf, lo_bf, o_bf, WfP, bfc, prelu_a, out);
}
